// SSF_SABlock_64647847739573
// MI455X (gfx1250) — matrix operand order repaired and checked by reading, not compiled, not run on hardware
//
#include <hip/hip_runtime.h>

typedef _Float16 v16h __attribute__((ext_vector_type(16)));
typedef float    v8f  __attribute__((ext_vector_type(8)));

static constexpr int kB     = 8;
static constexpr int kS     = 1024;
static constexpr int kH     = 768;
static constexpr int kHeads = 12;
static constexpr int kD     = 64;          // head dim
static constexpr int kM     = kB * kS;     // 8192 rows
static constexpr int kNqkv  = 3 * kH;      // 2304

// ---------------------------------------------------------------------------
// CDNA5 async global->LDS copy (ASYNCcnt-tracked), 16 bytes per lane.
// LDS destination address is taken from a VGPR; the generic pointer's low 32 bits are the
// LDS-space byte address (aperture occupies the high 32 bits).
// Passing the pointer through ptrtoint + dummy operand CAPTURES the LDS
// array so the "memory" clobber is seen as a may-write to it (otherwise the
// compiler deletes the downstream ds_loads as reads of undef memory).
// ---------------------------------------------------------------------------
static __device__ __forceinline__ void async_copy_b128(_Float16* ldsDst,
                                                       const void* gsrc) {
    unsigned ldsByteOff = (unsigned)(unsigned long long)ldsDst;   // low 32 bits
    unsigned long long ga = (unsigned long long)gsrc;
    asm volatile("global_load_async_to_lds_b128 %0, %1, off"
                 :: "v"(ldsByteOff), "v"(ga), "v"(ldsDst) : "memory");
}
static __device__ __forceinline__ void async_wait0() {
    asm volatile("s_wait_asynccnt 0" ::: "memory");
}

// ---------------------------------------------------------------------------
// WMMA fragment helpers (layouts per CDNA5 ISA 7.12.2, wave32)
// ---------------------------------------------------------------------------

// A matrix 16x32 f16, row-major source. Lane m=lane&15 holds row m.
// v16h elems 0..7 -> K = ksel*8 + 0..7 ; elems 8..15 -> K = ksel*8 + 16..23
static __device__ __forceinline__ v16h frag_a_load(const _Float16* p0, int ld, int lane) {
    const int m  = lane & 15;
    const int ks = lane >> 4;
    const _Float16* p = p0 + (size_t)m * ld + ks * 8;
    v16h a;
#pragma unroll
    for (int i = 0; i < 8; ++i) { a[i] = p[i]; a[i + 8] = p[i + 16]; }
    return a;
}

// B matrix 32x16 f16 from K-contiguous storage (B^T row-major: bt[n][k]).
// Lane n=lane&15 holds column n; elems 0..7 -> K = ksel*8 + 0..7 ; elems 8..15 -> K = ksel*8 + 16..23
static __device__ __forceinline__ v16h frag_b_load(const _Float16* p0, int ld, int lane) {
    const int n  = lane & 15;
    const int ks = lane >> 4;
    const _Float16* p = p0 + (size_t)n * ld + ks * 8;
    v16h b;
#pragma unroll
    for (int i = 0; i < 8; ++i) { b[i] = p[i]; b[i + 8] = p[i + 16]; }
    return b;
}

static __device__ __forceinline__ v8f wmma_f16(v16h a, v16h b, v8f c) {
    return __builtin_amdgcn_wmma_f32_16x16x32_f16(false, a, false, b, (short)0, c,
                                                  false, false);
}

// ---------------------------------------------------------------------------
// Conversion / transpose kernels
// ---------------------------------------------------------------------------
__global__ __launch_bounds__(256) void k_cvt_f16(const float* __restrict__ src,
                                                 _Float16* __restrict__ dst, int n) {
    for (int i = blockIdx.x * blockDim.x + threadIdx.x; i < n;
         i += gridDim.x * blockDim.x)
        dst[i] = (_Float16)src[i];
}

// src[rows][cols] (f32) -> dst[cols][rows] (f16)
__global__ __launch_bounds__(256) void k_transpose_f16(const float* __restrict__ src,
                                                       _Float16* __restrict__ dst,
                                                       int rows, int cols) {
    int n = rows * cols;
    for (int i = blockIdx.x * blockDim.x + threadIdx.x; i < n;
         i += gridDim.x * blockDim.x) {
        int c = i % cols, r = i / cols;
        dst[(size_t)c * rows + r] = (_Float16)src[i];
    }
}

// ---------------------------------------------------------------------------
// Stage 1: QKV projection.  One block = one 16-row panel x 8 column tiles.
// A panel (16x768 f16 = 24KB) staged into LDS once via async-to-LDS b128,
// shared by all 8 waves; B weight fragments stream from L2 (weights resident).
//   q,k : [b, head, s, d] row-major ; vT : [b, head, d, s]
// ---------------------------------------------------------------------------
static constexpr int kQkvNTB = (kNqkv / 64 + 7) / 8;   // 5 column-tile groups

__global__ __launch_bounds__(256) void k_qkv_gemm(const _Float16* __restrict__ xh,
                                                  const _Float16* __restrict__ wT,
                                                  const float* __restrict__ sc1,
                                                  const float* __restrict__ sh1,
                                                  _Float16* __restrict__ qb,
                                                  _Float16* __restrict__ kb,
                                                  _Float16* __restrict__ vT) {
    __shared__ _Float16 ldsA[16 * kH];               // 24KB
    const int lane  = threadIdx.x & 31;
    const int wslot = threadIdx.x >> 5;
    const int mt  = blockIdx.x / kQkvNTB;
    const int ntb = blockIdx.x % kQkvNTB;
    const int nt  = ntb * 8 + wslot;                 // this wave's column tile
    const int row0 = mt * 16;

    // ---- async-stage the contiguous 16x768 A panel: 6 x b128 per thread ----
    {
        const _Float16* src = xh + (size_t)row0 * kH;
#pragma unroll
        for (int it = 0; it < 6; ++it) {
            const int c = threadIdx.x + it * 256;    // 16-byte chunk index
            async_copy_b128(ldsA + c * 8, src + c * 8);
        }
        async_wait0();
        __syncthreads();
    }

    if (nt < kNqkv / 64) {
        const int col0 = nt * 64;
        v8f c0 = {}, c1 = {}, c2 = {}, c3 = {};
#pragma unroll 4
        for (int k0 = 0; k0 < kH; k0 += 32) {
            v16h a  = frag_a_load(ldsA + k0, kH, lane);         // ds_load_b128
            v16h b0 = frag_b_load(wT + (size_t)(col0 +  0) * kH + k0, kH, lane);
            v16h b1 = frag_b_load(wT + (size_t)(col0 + 16) * kH + k0, kH, lane);
            v16h b2 = frag_b_load(wT + (size_t)(col0 + 32) * kH + k0, kH, lane);
            v16h b3 = frag_b_load(wT + (size_t)(col0 + 48) * kH + k0, kH, lane);
            c0 = wmma_f16(a, b0, c0);
            c1 = wmma_f16(a, b1, c1);
            c2 = wmma_f16(a, b2, c2);
            c3 = wmma_f16(a, b3, c3);
        }

        // Epilogue: SSF affine then scatter into q / k / v^T (f16).
        const int n  = lane & 15;
        const int ks = lane >> 4;
        const int qsel = col0 / kH;                  // 0=q, 1=k, 2=v
        const int head = (col0 % kH) / kD;           // uniform per 64-wide tile
#pragma unroll
        for (int nb = 0; nb < 4; ++nb) {
            v8f c = (nb == 0) ? c0 : (nb == 1) ? c1 : (nb == 2) ? c2 : c3;
            const int col = col0 + nb * 16 + n;
            const float scale = sc1[col], shift = sh1[col];
            const int dd = nb * 16 + n;
#pragma unroll
            for (int i = 0; i < 8; ++i) {
                const int row = row0 + i + 8 * ks;   // C layout: M = i + 8*ksel
                const int bb = row >> 10, ss = row & (kS - 1);
                const float v = c[i] * scale + shift;
                const size_t bh = (size_t)(bb * kHeads + head);
                if (qsel == 0)      qb[(bh * kS + ss) * kD + dd] = (_Float16)v;
                else if (qsel == 1) kb[(bh * kS + ss) * kD + dd] = (_Float16)v;
                else                vT[(bh * kD + dd) * kS + ss] = (_Float16)v;
            }
        }
    }
}

// ---------------------------------------------------------------------------
// Stage 2: flash attention. One wave owns 16 query rows of one (b,head).
// Online softmax; P tile re-layout C->A through a 1KB per-wave LDS slab.
// ---------------------------------------------------------------------------
__global__ __launch_bounds__(256) void k_attn(const _Float16* __restrict__ qb,
                                              const _Float16* __restrict__ kb,
                                              const _Float16* __restrict__ vT,
                                              _Float16* __restrict__ oh) {
    __shared__ _Float16 lds[8 * 16 * 32];       // 8 waves x (16x32) P tile
    const int lane  = threadIdx.x & 31;
    const int wslot = threadIdx.x >> 5;
    _Float16* pslab = lds + wslot * 16 * 32;

    const int wave = blockIdx.x * 8 + wslot;
    const int qi = wave & 63;                   // query block (16 rows)
    const int bh = wave >> 6;                   // b*kHeads + head

    const _Float16* Q  = qb + ((size_t)bh * kS + qi * 16) * kD;
    const _Float16* K  = kb + (size_t)bh * kS * kD;
    const _Float16* Vt = vT + (size_t)bh * kD * kS;

    // Q fragments for d=0..31 and d=32..63, pre-scaled by 1/sqrt(64)
    v16h aq0 = frag_a_load(Q,      kD, lane);
    v16h aq1 = frag_a_load(Q + 32, kD, lane);
    const _Float16 qscale = (_Float16)0.125f;
#pragma unroll
    for (int i = 0; i < 16; ++i) { aq0[i] *= qscale; aq1[i] *= qscale; }

    float mrow[8], lrow[8];
#pragma unroll
    for (int i = 0; i < 8; ++i) { mrow[i] = -__builtin_inff(); lrow[i] = 0.0f; }
    v8f o0 = {}, o1 = {}, o2 = {}, o3 = {};

    const int n  = lane & 15;
    const int ks = lane >> 4;

    for (int j = 0; j < kS; j += 32) {          // 32 keys per step
        // ---- scores: S = (Q/8) K^T for two 16-key half tiles ----
        v8f s0 = {}, s1 = {};
        s0 = wmma_f16(aq0, frag_b_load(K + (size_t)(j)      * kD,      kD, lane), s0);
        s0 = wmma_f16(aq1, frag_b_load(K + (size_t)(j)      * kD + 32, kD, lane), s0);
        s1 = wmma_f16(aq0, frag_b_load(K + (size_t)(j + 16) * kD,      kD, lane), s1);
        s1 = wmma_f16(aq1, frag_b_load(K + (size_t)(j + 16) * kD + 32, kD, lane), s1);

        // ---- online softmax update (row stats across 16-lane half-wave) ----
#pragma unroll
        for (int i = 0; i < 8; ++i) {
            float rm = fmaxf(s0[i], s1[i]);
            rm = fmaxf(rm, __shfl_xor(rm, 1));
            rm = fmaxf(rm, __shfl_xor(rm, 2));
            rm = fmaxf(rm, __shfl_xor(rm, 4));
            rm = fmaxf(rm, __shfl_xor(rm, 8));
            const float mnew = fmaxf(mrow[i], rm);
            const float fac  = __expf(mrow[i] - mnew);
            const float p0   = __expf(s0[i] - mnew);
            const float p1   = __expf(s1[i] - mnew);
            float rs = p0 + p1;
            rs += __shfl_xor(rs, 1);
            rs += __shfl_xor(rs, 2);
            rs += __shfl_xor(rs, 4);
            rs += __shfl_xor(rs, 8);
            lrow[i] = lrow[i] * fac + rs;
            mrow[i] = mnew;
            o0[i] *= fac; o1[i] *= fac; o2[i] *= fac; o3[i] *= fac;
            // park P in LDS (C layout -> linear [row][key])
            pslab[(i + 8 * ks) * 32 + n]      = (_Float16)p0;
            pslab[(i + 8 * ks) * 32 + 16 + n] = (_Float16)p1;
        }
        asm volatile("s_wait_dscnt 0" ::: "memory");   // wave-local LDS RAW

        // ---- reload P as an A fragment (16x32) ----
        v16h ap;
        {
            const int m = lane & 15;
#pragma unroll
            for (int i = 0; i < 16; ++i) {
                const int kk = ks * 8 + (i & 7) + ((i >> 3) << 4);
                ap[i] = pslab[m * 32 + kk];
            }
        }

        // ---- O += P V  (V^T storage -> contiguous B fragments) ----
        o0 = wmma_f16(ap, frag_b_load(Vt + (size_t)0  * kS + j, kS, lane), o0);
        o1 = wmma_f16(ap, frag_b_load(Vt + (size_t)16 * kS + j, kS, lane), o1);
        o2 = wmma_f16(ap, frag_b_load(Vt + (size_t)32 * kS + j, kS, lane), o2);
        o3 = wmma_f16(ap, frag_b_load(Vt + (size_t)48 * kS + j, kS, lane), o3);
    }

    // ---- finalize: divide by l, store heads-interleaved [b, s, h*64+d] ----
    const int b = bh / kHeads, h = bh % kHeads;
#pragma unroll
    for (int i = 0; i < 8; ++i) {
        const float inv = 1.0f / lrow[i];
        const int row = qi * 16 + i + 8 * ks;
        _Float16* dst = oh + ((size_t)b * kS + row) * kH + h * kD + n;
        dst[0]  = (_Float16)(o0[i] * inv);
        dst[16] = (_Float16)(o1[i] * inv);
        dst[32] = (_Float16)(o2[i] * inv);
        dst[48] = (_Float16)(o3[i] * inv);
    }
}

// ---------------------------------------------------------------------------
// Stage 3: output projection with async-staged A panel (same scheme as QKV).
// oh[8192x768] @ WoutT + bout, then SSF2 -> f32 out
// ---------------------------------------------------------------------------
static constexpr int kOutNTB = (kH / 64 + 7) / 8;      // 2 column-tile groups

__global__ __launch_bounds__(256) void k_out_gemm(const _Float16* __restrict__ oh,
                                                  const _Float16* __restrict__ wT,
                                                  const float* __restrict__ bout,
                                                  const float* __restrict__ sc2,
                                                  const float* __restrict__ sh2,
                                                  float* __restrict__ out) {
    __shared__ _Float16 ldsA[16 * kH];               // 24KB
    const int lane  = threadIdx.x & 31;
    const int wslot = threadIdx.x >> 5;
    const int mt  = blockIdx.x / kOutNTB;
    const int ntb = blockIdx.x % kOutNTB;
    const int nt  = ntb * 8 + wslot;
    const int row0 = mt * 16;

    {
        const _Float16* src = oh + (size_t)row0 * kH;
#pragma unroll
        for (int it = 0; it < 6; ++it) {
            const int c = threadIdx.x + it * 256;
            async_copy_b128(ldsA + c * 8, src + c * 8);
        }
        async_wait0();
        __syncthreads();
    }

    if (nt < kH / 64) {
        const int col0 = nt * 64;
        v8f c0 = {}, c1 = {}, c2 = {}, c3 = {};
#pragma unroll 4
        for (int k0 = 0; k0 < kH; k0 += 32) {
            v16h a = frag_a_load(ldsA + k0, kH, lane);
            c0 = wmma_f16(a, frag_b_load(wT + (size_t)(col0 +  0) * kH + k0, kH, lane), c0);
            c1 = wmma_f16(a, frag_b_load(wT + (size_t)(col0 + 16) * kH + k0, kH, lane), c1);
            c2 = wmma_f16(a, frag_b_load(wT + (size_t)(col0 + 32) * kH + k0, kH, lane), c2);
            c3 = wmma_f16(a, frag_b_load(wT + (size_t)(col0 + 48) * kH + k0, kH, lane), c3);
        }

        const int n  = lane & 15;
        const int ks = lane >> 4;
#pragma unroll
        for (int nb = 0; nb < 4; ++nb) {
            v8f c = (nb == 0) ? c0 : (nb == 1) ? c1 : (nb == 2) ? c2 : c3;
            const int col = col0 + nb * 16 + n;
            const float bias = bout[col], scl = sc2[col], shf = sh2[col];
#pragma unroll
            for (int i = 0; i < 8; ++i) {
                const int row = row0 + i + 8 * ks;
                out[(size_t)row * kH + col] = (c[i] + bias) * scl + shf;
            }
        }
    }
}

// ---------------------------------------------------------------------------
// Host launcher
// ---------------------------------------------------------------------------
extern "C" void kernel_launch(void* const* d_in, const int* in_sizes, int n_in,
                              void* d_out, int out_size, void* d_ws, size_t ws_size,
                              hipStream_t stream) {
    (void)in_sizes; (void)n_in; (void)out_size; (void)ws_size;
    const float* x    = (const float*)d_in[0];
    const float* Wqkv = (const float*)d_in[1];
    const float* sc1  = (const float*)d_in[2];
    const float* sh1  = (const float*)d_in[3];
    const float* Wout = (const float*)d_in[4];
    const float* bo   = (const float*)d_in[5];
    const float* sc2  = (const float*)d_in[6];
    const float* sh2  = (const float*)d_in[7];
    float* out = (float*)d_out;

    char* ws = (char*)d_ws;
    size_t off = 0;
    auto carve = [&](size_t halves) -> _Float16* {
        _Float16* p = (_Float16*)(ws + off);
        off += (halves * sizeof(_Float16) + 255) & ~(size_t)255;
        return p;
    };
    _Float16* xh    = carve((size_t)kM * kH);               // x as f16
    _Float16* wqkvT = carve((size_t)kNqkv * kH);            // Wqkv^T f16
    _Float16* woutT = carve((size_t)kH * kH);               // Wout^T f16
    _Float16* qbuf  = carve((size_t)kB * kHeads * kS * kD); // [b,h,s,d]
    _Float16* kbuf  = carve((size_t)kB * kHeads * kS * kD); // [b,h,s,d]
    _Float16* vtb   = carve((size_t)kB * kHeads * kD * kS); // [b,h,d,s]
    _Float16* ohb   = carve((size_t)kM * kH);               // attn out f16

    k_cvt_f16<<<2048, 256, 0, stream>>>(x, xh, kM * kH);
    k_transpose_f16<<<2048, 256, 0, stream>>>(Wqkv, wqkvT, kH, kNqkv);
    k_transpose_f16<<<1024, 256, 0, stream>>>(Wout, woutT, kH, kH);

    {   // QKV GEMM: 512 row panels x 5 column-tile groups
        k_qkv_gemm<<<(kM / 16) * kQkvNTB, 256, 0, stream>>>(xh, wqkvT, sc1, sh1,
                                                            qbuf, kbuf, vtb);
    }
    {   // Attention: 8*12*(1024/16) = 6144 waves, 8 per block
        const int waves = kB * kHeads * (kS / 16);
        k_attn<<<(waves + 7) / 8, 256, 0, stream>>>(qbuf, kbuf, vtb, ohb);
    }
    {   // Output GEMM: 512 row panels x 2 column-tile groups
        k_out_gemm<<<(kM / 16) * kOutNTB, 256, 0, stream>>>(ohb, woutT, bo, sc2,
                                                            sh2, out);
    }
}
